// MultiHeadSelfAttention_83872121356645
// MI455X (gfx1250) — compile-verified, matrix operand order checked by reading, not run on hardware
//
#include <hip/hip_runtime.h>

// MI455X (gfx1250) multi-head attention forward.
// B=2, S=2048, D=1024, H=16, DK=64, ALPHA=0.5
// ~70 GFLOP vs ~135MB unique traffic -> compute bound -> bf16 WMMA everywhere.
// Flash attention stages K/V tiles into LDS with gfx1250 async loads
// (global_load_async_to_lds_b128 + s_wait_asynccnt), double buffered.

#define BB 2
#define SSQ 2048
#define DDIM 1024
#define HH 16
#define DKH 64
#define ALPHA_C 0.5f

typedef __attribute__((ext_vector_type(16))) __bf16 v16bf;
typedef __attribute__((ext_vector_type(8)))  float  v8f;
typedef __attribute__((ext_vector_type(4)))  float  v4f;
typedef __attribute__((ext_vector_type(4)))  int    v4i;

static __device__ __forceinline__ v8f wmma_bf16(v16bf a, v16bf b, v8f c) {
  // D = A(16x32 bf16) x B(32x16 bf16) + C(16x16 f32)
  return __builtin_amdgcn_wmma_f32_16x16x32_bf16(false, a, false, b, (short)0, c,
                                                 false, false);
}

static __device__ __forceinline__ v8f vzero8() {
  v8f z = {0.f, 0.f, 0.f, 0.f, 0.f, 0.f, 0.f, 0.f};
  return z;
}

// Async 16B global->LDS copy (per-lane addresses; tracked by ASYNCcnt).
// LDS aperture maps addr[31:0] directly to the wave-relative LDS address
// (ISA 10.2), so the low 32 bits of the generic shared pointer are the
// VDST LDS address the instruction wants.
static __device__ __forceinline__ void async_copy16(void* lds_dst,
                                                    const void* gsrc) {
  unsigned loff = (unsigned)(unsigned long long)(uintptr_t)lds_dst;
  unsigned long long g = (unsigned long long)(uintptr_t)gsrc;
  asm volatile("global_load_async_to_lds_b128 %0, %1, off"
               :: "v"(loff), "v"(g)
               : "memory");
}
#define S_WAIT_ASYNCCNT(n) asm volatile("s_wait_asynccnt " #n ::: "memory")

// ---------------------------------------------------------------- convert ---
__global__ void __launch_bounds__(256)
cvt_f32_bf16(const float* __restrict__ src, __bf16* __restrict__ dst, int n) {
  int i = (blockIdx.x * 256 + threadIdx.x) * 4;
  if (i + 3 < n) {
    v4f x = *(const v4f*)(src + i);
    dst[i + 0] = (__bf16)x[0];
    dst[i + 1] = (__bf16)x[1];
    dst[i + 2] = (__bf16)x[2];
    dst[i + 3] = (__bf16)x[3];
  }
}

// ------------------------------------------------------- projection GEMMs ---
// Y[m,n] = sum_k X[m,k]*W[n,k] + bias[n];  X:[4096,1024] bf16, W:[1024,1024] bf16
// TRANSV==0 -> Yh[b,h,s,d]; TRANSV==1 -> Yh[b,h,d,s] (for the PV fragments).
// Wave tile 32x32 (4 C tiles); NT layout -> every fragment is one 32B load.
template <int TRANSV>
__global__ void __launch_bounds__(256)
proj_gemm(const __bf16* __restrict__ X, const __bf16* __restrict__ W,
          const float* __restrict__ bias, __bf16* __restrict__ Yh) {
  const int lane = threadIdx.x & 31;
  const int wid  = threadIdx.x >> 5;
  const int m0 = blockIdx.y * 128 + (wid & 3) * 32;
  const int n0 = blockIdx.x * 64  + (wid >> 2) * 32;
  const int lr = lane & 15;   // row/col within tile
  const int hi = lane >> 4;   // K-half of fragment

  v8f c00 = vzero8(), c01 = vzero8(), c10 = vzero8(), c11 = vzero8();

  for (int k = 0; k < DDIM; k += 32) {
    v16bf a0 = *(const v16bf*)(X + (size_t)(m0 +      lr) * DDIM + k + hi * 16);
    v16bf a1 = *(const v16bf*)(X + (size_t)(m0 + 16 + lr) * DDIM + k + hi * 16);
    v16bf b0 = *(const v16bf*)(W + (size_t)(n0 +      lr) * DDIM + k + hi * 16);
    v16bf b1 = *(const v16bf*)(W + (size_t)(n0 + 16 + lr) * DDIM + k + hi * 16);
    c00 = wmma_bf16(a0, b0, c00);
    c01 = wmma_bf16(a0, b1, c01);
    c10 = wmma_bf16(a1, b0, c10);
    c11 = wmma_bf16(a1, b1, c11);
  }

  v8f cs[2][2];
  cs[0][0] = c00; cs[0][1] = c01; cs[1][0] = c10; cs[1][1] = c11;
#pragma unroll
  for (int mi = 0; mi < 2; ++mi) {
#pragma unroll
    for (int ni = 0; ni < 2; ++ni) {
      const int ng = n0 + ni * 16 + lr;        // C layout: col = lane&15
      const float bv = bias[ng];
      const int hd = ng >> 6, d = ng & 63;
#pragma unroll
      for (int v = 0; v < 8; ++v) {            // C layout: row = vgpr + 8*hi
        const int mg = m0 + mi * 16 + hi * 8 + v;
        const int b = mg >> 11, s = mg & 2047;
        const float val = cs[mi][ni][v] + bv;
        size_t idx;
        if (TRANSV == 0) idx = (((size_t)(b * HH + hd)) * SSQ + s) * DKH + d;
        else             idx = (((size_t)(b * HH + hd)) * DKH + d) * SSQ + s;
        Yh[idx] = (__bf16)val;
      }
    }
  }
}

// -------------------------------------------------------- flash attention ---
// Block = 8 waves covering 8 query tiles of one (b,h). K (32x64) and V^T
// (64x32) tiles are staged into LDS once per block via async loads, double
// buffered. Each wave computes S^T = K*Q^T so that:
//  - K / V^T A-frags and Q B-frags are contiguous 32B (ds_/global_) loads
//  - per-query softmax state is a per-lane scalar; reductions = 1 shfl_xor
//  - per-lane mask elements are contiguous -> int4/float4 loads
// O^T = V^T*P accumulated in 4 C tiles; P^T C-tiles -> P B-frag via 16
// half-wave shuffles (no LDS round-trip).
__global__ void __launch_bounds__(256)
attn_fwd(const __bf16* __restrict__ Qh,   // [B,H,S,DK]
         const __bf16* __restrict__ Kh,   // [B,H,S,DK]
         const __bf16* __restrict__ VhT,  // [B,H,DK,S]
         const int*   __restrict__ amask, // [S,S]
         const float* __restrict__ cmask, // [S,S]
         float* __restrict__ attn_out,    // [B,H,S,DK] (f32, second output)
         __bf16* __restrict__ attn_bf) {  // [B,H,S,DK] bf16 copy for out GEMM
  __shared__ __bf16 lk[2][32 * DKH];      // [keys 32][d 64]
  __shared__ __bf16 lv[2][DKH * 32];      // [d 64][keys 32]

  const int tid  = threadIdx.x;
  const int lane = tid & 31;
  const int wid  = tid >> 5;
  const int bh = blockIdx.x >> 4;               // B*H = 32, 16 blocks each
  const int qbase = (blockIdx.x & 15) * 128 + wid * 16;
  const int lr = lane & 15, hi = lane >> 4;

  const __bf16* Qb = Qh  + (size_t)bh * SSQ * DKH;
  const __bf16* Kb = Kh  + (size_t)bh * SSQ * DKH;
  const __bf16* Vb = VhT + (size_t)bh * DKH * SSQ;

  // Per-thread staging slots: one 16B chunk of each tile per thread.
  // K tile = contiguous 4KB at Kb + j*64.  V tile = 64 rows (stride S) x 64B.
  __bf16* lk_dst0 = &lk[0][tid * 8];
  __bf16* lk_dst1 = &lk[1][tid * 8];
  __bf16* lv_dst0 = &lv[0][(tid >> 2) * 32 + (tid & 3) * 8];
  __bf16* lv_dst1 = &lv[1][(tid >> 2) * 32 + (tid & 3) * 8];
  const __bf16* vrow = Vb + (size_t)(tid >> 2) * SSQ + (tid & 3) * 8;

  // Q B-fragments (d 0..31 and 32..63), resident for the whole pass.
  v16bf qf0 = *(const v16bf*)(Qb + (size_t)(qbase + lr) * DKH +  0 + hi * 16);
  v16bf qf1 = *(const v16bf*)(Qb + (size_t)(qbase + lr) * DKH + 32 + hi * 16);

  v8f o[4];
#pragma unroll
  for (int t = 0; t < 4; ++t) o[t] = vzero8();
  float mrun = -3.0e38f, lrun = 0.f;
  const float inv_sqrt_dk = 0.125f;
  const int qrow = qbase + lr;

  // Prologue: fill buffer 0 with keys 0..31.
  async_copy16(lk_dst0, Kb + tid * 8);
  async_copy16(lv_dst0, vrow);

  for (int j = 0; j < SSQ; j += 32) {
    const int buf = (j >> 5) & 1;
    if (j + 32 < SSQ) {  // prefetch next tile into the other buffer
      __bf16* kd = buf ? lk_dst0 : lk_dst1;
      __bf16* vd = buf ? lv_dst0 : lv_dst1;
      async_copy16(kd, Kb + (size_t)(j + 32) * DKH + tid * 8);
      async_copy16(vd, vrow + j + 32);
      S_WAIT_ASYNCCNT(0x2);     // current buffer's 2 loads retired (in-order)
    } else {
      S_WAIT_ASYNCCNT(0x0);
    }
    __syncthreads();
    const __bf16* lkb = lk[buf];
    const __bf16* lvb = lv[buf];

    // S^T tiles: t0 = keys j..j+15, t1 = keys j+16..j+31 (rows), queries cols.
    v16bf ka0 = *(const v16bf*)(lkb + (     lr) * DKH +  0 + hi * 16);
    v16bf ka1 = *(const v16bf*)(lkb + (     lr) * DKH + 32 + hi * 16);
    v16bf kb0 = *(const v16bf*)(lkb + (16 + lr) * DKH +  0 + hi * 16);
    v16bf kb1 = *(const v16bf*)(lkb + (16 + lr) * DKH + 32 + hi * 16);
    v8f t0 = vzero8(), t1 = vzero8();
    t0 = wmma_bf16(ka0, qf0, t0);
    t0 = wmma_bf16(ka1, qf1, t0);
    t1 = wmma_bf16(kb0, qf0, t1);
    t1 = wmma_bf16(kb1, qf1, t1);

    // Masks: lane holds query `qrow`, keys j+8*hi+v (t0) / j+16+8*hi+v (t1).
    const int*   amr = amask + (size_t)qrow * SSQ + j;
    const float* cmr = cmask + (size_t)qrow * SSQ + j;
    v4i a0v = *(const v4i*)(amr + 8 * hi);
    v4i a1v = *(const v4i*)(amr + 8 * hi + 4);
    v4i a2v = *(const v4i*)(amr + 16 + 8 * hi);
    v4i a3v = *(const v4i*)(amr + 16 + 8 * hi + 4);
    v4f c0v = *(const v4f*)(cmr + 8 * hi);
    v4f c1v = *(const v4f*)(cmr + 8 * hi + 4);
    v4f c2v = *(const v4f*)(cmr + 16 + 8 * hi);
    v4f c3v = *(const v4f*)(cmr + 16 + 8 * hi + 4);

    float p0[8], p1[8];
    float mloc = -3.0e38f;
#pragma unroll
    for (int v = 0; v < 8; ++v) {
      const int   am0 = (v < 4) ? a0v[v & 3] : a1v[v & 3];
      const int   am1 = (v < 4) ? a2v[v & 3] : a3v[v & 3];
      const float cm0 = (v < 4) ? c0v[v & 3] : c1v[v & 3];
      const float cm1 = (v < 4) ? c2v[v & 3] : c3v[v & 3];
      float s0 = t0[v] * inv_sqrt_dk;
      float s1 = t1[v] * inv_sqrt_dk;
      s0 = (am0 == 1) ? -1.0e9f : s0;
      s1 = (am1 == 1) ? -1.0e9f : s1;
      s0 *= (1.0f + ALPHA_C * cm0);
      s1 *= (1.0f + ALPHA_C * cm1);
      p0[v] = s0;
      p1[v] = s1;
      mloc = fmaxf(mloc, fmaxf(s0, s1));
    }
    mloc = fmaxf(mloc, __shfl_xor(mloc, 16));
    const float mnew = fmaxf(mrun, mloc);
    const float corr = __expf(mrun - mnew);
    float psum = 0.f;
#pragma unroll
    for (int v = 0; v < 8; ++v) {
      p0[v] = __expf(p0[v] - mnew);
      p1[v] = __expf(p1[v] - mnew);
      psum += p0[v] + p1[v];
    }
    psum += __shfl_xor(psum, 16);
    lrun = lrun * corr + psum;
    mrun = mnew;
#pragma unroll
    for (int t = 0; t < 4; ++t)
#pragma unroll
      for (int v = 0; v < 8; ++v) o[t][v] *= corr;

    // P^T C-tiles -> P B-fragment (32 keys x 16 queries): half-wave exchange.
    v16bf pf;
#pragma unroll
    for (int v = 0; v < 8; ++v) {
      const float s0 = __shfl_xor(p0[v], 16);
      const float s1 = __shfl_xor(p1[v], 16);
      const float first = hi ? s1 : p0[v];  // keys 16*hi + v
      const float last  = hi ? p1[v] : s0;  // keys 16*hi + 8 + v
      pf[v]     = (__bf16)first;
      pf[v + 8] = (__bf16)last;
    }

    // O^T += V^T(d-block, keys j..j+31) * P
#pragma unroll
    for (int t = 0; t < 4; ++t) {
      v16bf vf = *(const v16bf*)(lvb + (t * 16 + lr) * 32 + hi * 16);
      o[t] = wmma_bf16(vf, pf, o[t]);
    }
    __syncthreads();  // everyone done with buf before next iter overwrites it
  }

  const float linv = 1.0f / lrun;
  float*  ao = attn_out + (size_t)bh * SSQ * DKH;
  __bf16* ab = attn_bf  + (size_t)bh * SSQ * DKH;
#pragma unroll
  for (int t = 0; t < 4; ++t) {
#pragma unroll
    for (int v = 0; v < 8; ++v) {
      const float val = o[t][v] * linv;
      const int dg = t * 16 + hi * 8 + v;          // O^T: row=d, col=query
      const size_t idx = (size_t)(qbase + lr) * DKH + dg;
      ao[idx] = val;
      ab[idx] = (__bf16)val;
    }
  }
}

// ---------------------------------------------------------- output GEMM -----
// out[m,n] = sum_k attn_bf[m,k]*Wo[n,k] + bo[n]; attn_bf stored [B,H,S,DK] so
// a 16-wide K chunk stays inside one head's d range -> contiguous fragment.
__global__ void __launch_bounds__(256)
out_gemm(const __bf16* __restrict__ Abf, const __bf16* __restrict__ W,
         const float* __restrict__ bias, float* __restrict__ out) {
  const int lane = threadIdx.x & 31;
  const int wid  = threadIdx.x >> 5;
  const int m0 = blockIdx.y * 128 + (wid & 3) * 32;
  const int n0 = blockIdx.x * 64  + (wid >> 2) * 32;
  const int lr = lane & 15;
  const int hi = lane >> 4;

  const int mg0 = m0 + lr,      b0i = mg0 >> 11, s0i = mg0 & 2047;
  const int mg1 = m0 + 16 + lr, b1i = mg1 >> 11, s1i = mg1 & 2047;

  v8f c00 = vzero8(), c01 = vzero8(), c10 = vzero8(), c11 = vzero8();
  for (int k = 0; k < DDIM; k += 32) {
    const int kb = k + hi * 16;
    const int h = kb >> 6, d = kb & 63;
    v16bf a0 = *(const v16bf*)(Abf + (((size_t)(b0i * HH + h)) * SSQ + s0i) * DKH + d);
    v16bf a1 = *(const v16bf*)(Abf + (((size_t)(b1i * HH + h)) * SSQ + s1i) * DKH + d);
    v16bf w0 = *(const v16bf*)(W + (size_t)(n0 +      lr) * DDIM + kb);
    v16bf w1 = *(const v16bf*)(W + (size_t)(n0 + 16 + lr) * DDIM + kb);
    c00 = wmma_bf16(a0, w0, c00);
    c01 = wmma_bf16(a0, w1, c01);
    c10 = wmma_bf16(a1, w0, c10);
    c11 = wmma_bf16(a1, w1, c11);
  }

  v8f cs[2][2];
  cs[0][0] = c00; cs[0][1] = c01; cs[1][0] = c10; cs[1][1] = c11;
#pragma unroll
  for (int mi = 0; mi < 2; ++mi) {
#pragma unroll
    for (int ni = 0; ni < 2; ++ni) {
      const int ng = n0 + ni * 16 + lr;
      const float bv = bias[ng];
#pragma unroll
      for (int v = 0; v < 8; ++v) {
        const int mg = m0 + mi * 16 + hi * 8 + v;
        out[(size_t)mg * DDIM + ng] = cs[mi][ni][v] + bv;
      }
    }
  }
}

// -------------------------------------------------------------- launcher ----
extern "C" void kernel_launch(void* const* d_in, const int* in_sizes, int n_in,
                              void* d_out, int out_size, void* d_ws, size_t ws_size,
                              hipStream_t stream) {
  (void)in_sizes; (void)n_in; (void)out_size; (void)ws_size;
  const float* q  = (const float*)d_in[0];
  const float* k  = (const float*)d_in[1];
  const float* v  = (const float*)d_in[2];
  const int*   am = (const int*)d_in[3];
  const float* cm = (const float*)d_in[4];
  const float* Wq = (const float*)d_in[5];  const float* bq = (const float*)d_in[6];
  const float* Wk = (const float*)d_in[7];  const float* bk = (const float*)d_in[8];
  const float* Wv = (const float*)d_in[9];  const float* bv = (const float*)d_in[10];
  const float* Wo = (const float*)d_in[11]; const float* bo = (const float*)d_in[12];

  float* out  = (float*)d_out;                           // [B,S,D]
  float* attn = out + (size_t)BB * SSQ * DDIM;           // [B,H,S,DK]

  const size_t NX = (size_t)BB * SSQ * DDIM;  // 4M elements
  const size_t NW = (size_t)DDIM * DDIM;      // 1M elements
  __bf16* w   = (__bf16*)d_ws;                // 64 MiB bf16 staging total
  __bf16* xq  = w;        __bf16* xk  = xq + NX;  __bf16* xv  = xk + NX;
  __bf16* wq  = xv + NX;  __bf16* wk  = wq + NW;  __bf16* wv  = wk + NW;
  __bf16* wo  = wv + NW;
  __bf16* qh  = wo + NW;  __bf16* kh  = qh + NX;  __bf16* vhT = kh + NX;
  __bf16* abf = vhT + NX;

  // 1) f32 -> bf16 staging
  cvt_f32_bf16<<<(int)(NX / 1024), 256, 0, stream>>>(q,  xq, (int)NX);
  cvt_f32_bf16<<<(int)(NX / 1024), 256, 0, stream>>>(k,  xk, (int)NX);
  cvt_f32_bf16<<<(int)(NX / 1024), 256, 0, stream>>>(v,  xv, (int)NX);
  cvt_f32_bf16<<<(int)(NW / 1024), 256, 0, stream>>>(Wq, wq, (int)NW);
  cvt_f32_bf16<<<(int)(NW / 1024), 256, 0, stream>>>(Wk, wk, (int)NW);
  cvt_f32_bf16<<<(int)(NW / 1024), 256, 0, stream>>>(Wv, wv, (int)NW);
  cvt_f32_bf16<<<(int)(NW / 1024), 256, 0, stream>>>(Wo, wo, (int)NW);

  // 2) projections (V stored per-head transposed for the PV WMMA fragments)
  dim3 grid(DDIM / 64, (BB * SSQ) / 128), blk(256);
  proj_gemm<0><<<grid, blk, 0, stream>>>(xq, wq, bq, qh);
  proj_gemm<0><<<grid, blk, 0, stream>>>(xk, wk, bk, kh);
  proj_gemm<1><<<grid, blk, 0, stream>>>(xv, wv, bv, vhT);

  // 3) flash attention: 512 blocks x 8 waves (one (b,h) + 128 queries each)
  attn_fwd<<<512, 256, 0, stream>>>(qh, kh, vhT, am, cm, attn, abf);

  // 4) output projection
  out_gemm<<<grid, blk, 0, stream>>>(abf, wo, bo, out);
}
